// RCOS_65747359367739
// MI455X (gfx1250) — matrix operand order repaired and checked by reading, not compiled, not run on hardware
//
#include <hip/hip_runtime.h>

typedef __attribute__((ext_vector_type(16))) __bf16 v16bf;
typedef __attribute__((ext_vector_type(8)))  __bf16 v8bf;
typedef __attribute__((ext_vector_type(4)))  __bf16 v4bf;
typedef __attribute__((ext_vector_type(8)))  float  v8f;
typedef __attribute__((ext_vector_type(4)))  float  f32x4;
typedef __attribute__((ext_vector_type(4)))  unsigned int u32x4;
typedef __attribute__((ext_vector_type(8)))  int    i32x8;
typedef __attribute__((ext_vector_type(4)))  int    i32x4;

#define NH   8      // heads (== waves per block)
#define NN   1024   // sequence length
#define DD   128    // head dim
#define TM   32     // m-tile (K of the PV GEMM)
#define TN   16     // n rows per workgroup

// LDS layout (bytes)
//   stage : [H][TM][D]  f32   = 131072  (TDM staging, raw fp32 tile)
//   k_row : [H][TM][D]  bf16  = 65536   (row-major K/V tile, E-GEMM B operand)
//   v_col : [H][D][TM]  bf16  = 65536   (d-major   K/V tile, PV-GEMM B operand)
//   e_buf : [H][TN][TM] f32   = 16384   (energies for cross-head softmax)
//   a_buf : [H][TN][TM] bf16  =  8192   (softmaxed attention, A-operand-permuted m)
#define STAGE_OFF 0
#define KROW_OFF  131072
#define VCOL_OFF  196608
#define EBUF_OFF  262144
#define ABUF_OFF  278528
#define SMEM_BYTES 286720

// permute m so the PV A-operand per-lane 16 K-values are contiguous in LDS
__device__ __forceinline__ int mperm(int m) {
    return (m & 7) | ((m & 8) << 1) | ((m & 16) >> 1);
}

// Issue one TDM load of the fp32 K/V tile kv[b, 0:8, m0:m0+32, 0:128] -> LDS stage.
// D# built per CDNA5 ISA:
//   3D tile: x=d (128), y=m (32, stride 128), z=h (8, stride N*D=131072), 4B elems.
// This toolchain exposes the 6-arg builtin: (v4u, v8i, v4i, v4i, v8i, i32 cpol).
__device__ __forceinline__ void tdm_load_tile(const float* gbase, unsigned lds_off) {
    unsigned long long ga = (unsigned long long)(uintptr_t)gbase;
    u32x4 g0;
    g0[0] = 1u;                                   // count=1, flags 0, gather off
    g0[1] = lds_off;                              // lds_addr (bytes)
    g0[2] = (unsigned)ga;                         // global_addr[31:0]
    g0[3] = (unsigned)((ga >> 32) & 0x01FFFFFFu)  // global_addr[56:32]
          | (2u << 30);                           // type=2 ("image")
    i32x8 g1;
    g1[0] = (int)(2u << 16);                      // data_size=2 -> 4 bytes
    g1[1] = (int)(128u << 16);                    // tensor_dim0 = 128 (d)
    g1[2] = (int)(1024u << 16);                   // tensor_dim1 = 1024 (m extent)
    g1[3] = (int)(128u << 16);                    // tile_dim0 = 128
    g1[4] = (int)(32u | (8u << 16));              // tile_dim1 = 32, tile_dim2 = 8
    g1[5] = 128;                                  // tensor_dim0_stride = 128 (row)
    g1[6] = 0;                                    // dim1_stride[15:0] = 0 (131072)
    g1[7] = 2;                                    // dim1_stride[47:16] = 2 -> 0x20000
    i32x4 g2;
    g2[0] = 8;                                    // tensor_dim2 = 8 (heads)
    g2[1] = 1;                                    // tensor_dim3 (unused)
    g2[2] = 0;                                    // tensor_dim2_stride (unused, 3D)
    g2[3] = 0;                                    // tile_dim3 = 0 (unused)
    i32x4 g3 = {0, 0, 0, 0};
    i32x8 g4 = {0, 0, 0, 0, 0, 0, 0, 0};          // extra operand (zero-filled)
    __builtin_amdgcn_tensor_load_to_lds(g0, g1, g2, g3, g4, 0);
}

template<bool QBF16, bool RESID>
__global__ __launch_bounds__(256, 1)
void attn_stage(const void* __restrict__ qp,
                const float* __restrict__ kv,
                void* __restrict__ outp)
{
    extern __shared__ char smem[];
    float*  stage = (float*) (smem + STAGE_OFF);
    __bf16* k_row = (__bf16*)(smem + KROW_OFF);
    __bf16* v_col = (__bf16*)(smem + VCOL_OFF);
    float*  e_buf = (float*) (smem + EBUF_OFF);
    __bf16* a_buf = (__bf16*)(smem + ABUF_OFF);

    const int b    = blockIdx.y;
    const int n0   = blockIdx.x * TN;
    const int t    = threadIdx.x;
    const int h    = t >> 5;        // wave id == head id
    const int lane = t & 31;
    const int c    = lane & 15;     // M (A-operand) / N (B,D operands) index
    const int hi   = lane >> 4;
    const int k0   = hi * 8;        // A-operand interleave base
    const int kb   = hi * 16;       // A-operand (permuted a_buf) load base

    const float inv_scale = 0.08838834764831845f; // 1/sqrt(128)
    const unsigned stage_lds = (unsigned)(uintptr_t)(void*)stage;
    const float* kv_b = kv + ((size_t)b * NH) * NN * DD;

    // kick off the TDM for tile 0 (wave 0 only; TENSORcnt is per-wave)
    if (h == 0) tdm_load_tile(kv_b, stage_lds);

    // ---- resident Q tile for this head, A-operand layout, 4 chunks of K=32 ----
    v16bf qa[4];
    {
        const int n = c;
        if (QBF16) {
            const __bf16* q = (const __bf16*)qp + (((size_t)(b*NH + h))*NN + (n0 + n))*DD;
            #pragma unroll
            for (int ch = 0; ch < 4; ++ch) {
                v8bf lo = *(const v8bf*)(q + ch*32 + k0);
                v8bf hv = *(const v8bf*)(q + ch*32 + 16 + k0);
                #pragma unroll
                for (int i = 0; i < 8; ++i) { qa[ch][i] = lo[i]; qa[ch][i+8] = hv[i]; }
            }
        } else {
            const float* q = (const float*)qp + (((size_t)(b*NH + h))*NN + (n0 + n))*DD;
            #pragma unroll
            for (int ch = 0; ch < 4; ++ch) {
                f32x4 a0 = *(const f32x4*)(q + ch*32 + k0);
                f32x4 a1 = *(const f32x4*)(q + ch*32 + k0 + 4);
                f32x4 b0 = *(const f32x4*)(q + ch*32 + 16 + k0);
                f32x4 b1 = *(const f32x4*)(q + ch*32 + 16 + k0 + 4);
                #pragma unroll
                for (int i = 0; i < 4; ++i) {
                    qa[ch][i]    = (__bf16)a0[i];
                    qa[ch][i+4]  = (__bf16)a1[i];
                    qa[ch][i+8]  = (__bf16)b0[i];
                    qa[ch][i+12] = (__bf16)b1[i];
                }
            }
        }
    }

    v8f x_acc[8];
    #pragma unroll
    for (int i = 0; i < 8; ++i) x_acc[i] = (v8f){};

    for (int m0 = 0; m0 < NN; m0 += TM) {
        // ---- tile i has landed in staging (TDM tracked by wave 0) ----
        if (h == 0) __builtin_amdgcn_s_wait_tensorcnt(0);
        __syncthreads();

        // ---- convert staging fp32 -> dual bf16 layouts ----
        {
            const int hl    = t >> 5;
            const int dbase = (t & 31) * 4;
            const float* stg = stage + (hl*TM)*DD + dbase;
            #pragma unroll 4
            for (int m = 0; m < TM; ++m) {
                f32x4 v = *(const f32x4*)(stg + m * DD);
                v4bf pk;
                #pragma unroll
                for (int j = 0; j < 4; ++j) pk[j] = (__bf16)v[j];
                *(v4bf*)(k_row + ((hl*TM + m)*DD + dbase)) = pk;
                #pragma unroll
                for (int j = 0; j < 4; ++j)
                    v_col[(hl*DD + dbase + j)*TM + m] = pk[j];
            }
        }
        __syncthreads();   // staging fully consumed; bf16 tiles visible

        // ---- overlap: DMA tile i+1 into staging while we do the WMMA work ----
        if (h == 0 && (m0 + TM) < NN)
            tdm_load_tile(kv_b + (size_t)(m0 + TM) * DD, stage_lds);

        // ---- E = Q . K^T (this head), two 16-wide m halves ----
        #pragma unroll
        for (int mh = 0; mh < 2; ++mh) {
            v8f e = (v8f){};
            const int m = mh*16 + c;                       // B operand: N = m
            const __bf16* krow = k_row + (h*TM + m)*DD;
            #pragma unroll
            for (int ch = 0; ch < 4; ++ch) {
                v16bf bb = __builtin_shufflevector(*(const v8bf*)(krow + ch*32 + k0), *(const v8bf*)(krow + ch*32 + 16 + k0), 0,1,2,3,4,5,6,7,8,9,10,11,12,13,14,15); // interleaved K (k0, 16+k0)
                e = __builtin_amdgcn_wmma_f32_16x16x32_bf16(false, qa[ch], false, bb,
                                                            (short)0, e, false, false);
            }
            #pragma unroll
            for (int r = 0; r < 8; ++r)   // D layout: lane=col m, VGPR r -> row r+8*hi
                e_buf[(h*TN + (r + 8*hi))*TM + (mh*16 + c)] = e[r] * inv_scale;
        }
        __syncthreads();

        // ---- softmax over the HEAD axis, pointwise per (n,m) ----
        {
            const int n     = t >> 4;
            const int mbase = (t & 15) * 2;
            #pragma unroll
            for (int p = 0; p < 2; ++p) {
                const int m  = mbase + p;
                const int mp = mperm(m);
                float ev[NH];
                float mx = -1e30f;
                #pragma unroll
                for (int hh = 0; hh < NH; ++hh) {
                    ev[hh] = e_buf[(hh*TN + n)*TM + m];
                    mx = fmaxf(mx, ev[hh]);
                }
                float s = 0.f;
                #pragma unroll
                for (int hh = 0; hh < NH; ++hh) { ev[hh] = __expf(ev[hh] - mx); s += ev[hh]; }
                const float rs = 1.0f / s;
                #pragma unroll
                for (int hh = 0; hh < NH; ++hh)
                    a_buf[(hh*TN + n)*TM + mp] = (__bf16)(ev[hh] * rs);
            }
        }
        __syncthreads();

        // ---- X += A . V (this head), K = 32 = m-tile ----
        {
            // A operand: M = n = c; permuted a_buf row makes this one 32B load
            v16bf aa = *(const v16bf*)(a_buf + (h*TN + c)*TM + kb);
            #pragma unroll
            for (int ds = 0; ds < 8; ++ds) {
                const int dcol = ds*16 + c;               // B operand: N = dcol
                v16bf bb = __builtin_shufflevector(*(const v8bf*)(v_col + (h*DD + dcol)*TM + k0), *(const v8bf*)(v_col + (h*DD + dcol)*TM + 16 + k0), 0,1,2,3,4,5,6,7,8,9,10,11,12,13,14,15);
                x_acc[ds] = __builtin_amdgcn_wmma_f32_16x16x32_bf16(false, aa, false, bb,
                                                                    (short)0, x_acc[ds],
                                                                    false, false);
            }
        }
        __syncthreads();   // PV reads done before next convert overwrites bf16 tiles
    }

    // ---- epilogue: D layout lane=col c, VGPR r -> row r+8*hi ----
    if (RESID) {
        float* out = (float*)outp;
        const float* resid = kv;  // stage 2: kv == sal == residual
        #pragma unroll
        for (int ds = 0; ds < 8; ++ds) {
            #pragma unroll
            for (int r = 0; r < 8; ++r) {
                const int n = r + 8*hi;
                const size_t idx = (((size_t)(b*NH + h))*NN + (n0 + n))*DD + ds*16 + c;
                out[idx] = resid[idx] + x_acc[ds][r];
            }
        }
    } else {
        __bf16* out = (__bf16*)outp;
        #pragma unroll
        for (int ds = 0; ds < 8; ++ds) {
            #pragma unroll
            for (int r = 0; r < 8; ++r) {
                const int n = r + 8*hi;
                const size_t idx = (((size_t)(b*NH + h))*NN + (n0 + n))*DD + ds*16 + c;
                out[idx] = (__bf16)x_acc[ds][r];
            }
        }
    }
}

extern "C" void kernel_launch(void* const* d_in, const int* in_sizes, int n_in,
                              void* d_out, int out_size, void* d_ws, size_t ws_size,
                              hipStream_t stream) {
    const float* sal = (const float*)d_in[0];
    const float* si  = (const float*)d_in[1];
    float*  out   = (float*)d_out;
    __bf16* xself = (__bf16*)d_ws;          // 8*8*1024*128 bf16 = 16 MB scratch

    dim3 grid(NN / TN, 8);                  // 64 n-tiles x B=8
    dim3 block(256);                        // 8 waves == 8 heads

    // Stage 1: Q=K=V=si (fp32 in), softmax over H, write x_self as bf16.
    attn_stage<false, false><<<grid, block, SMEM_BYTES, stream>>>(si, si, xself);
    // Stage 2: Q=x_self (bf16), K=V=sal, softmax over H, out = sal + A.V (fp32).
    attn_stage<true, true><<<grid, block, SMEM_BYTES, stream>>>(xself, sal, out);
}
